// Cross_Att_17197049053535
// MI455X (gfx1250) — matrix operand order repaired and checked by reading, not compiled, not run on hardware
//
#include <hip/hip_runtime.h>
#include <stdint.h>

// ---------------- problem constants ----------------
#define BATCH 16
#define CCH   256
#define NPIX  4096      // 64*64
#define NH    8
#define DH    32
#define NG    32
#define EPSV  1e-5f

typedef __attribute__((ext_vector_type(16))) __bf16 v16bf;
typedef __attribute__((ext_vector_type(8)))  float  v8f;

union Frag { v16bf v; uint4 u[2]; };

__device__ __forceinline__ unsigned short f2bf(float f) {
  unsigned u = __float_as_uint(f);
  u += 0x7fffu + ((u >> 16) & 1u);
  return (unsigned short)(u >> 16);
}
__device__ __forceinline__ float bf2f(unsigned short h) {
  return __uint_as_float(((unsigned)h) << 16);
}
__device__ __forceinline__ uint4 pack8(v8f a) {
  uint4 o;
  o.x = (unsigned)f2bf(a[0]) | ((unsigned)f2bf(a[1]) << 16);
  o.y = (unsigned)f2bf(a[2]) | ((unsigned)f2bf(a[3]) << 16);
  o.z = (unsigned)f2bf(a[4]) | ((unsigned)f2bf(a[5]) << 16);
  o.w = (unsigned)f2bf(a[6]) | ((unsigned)f2bf(a[7]) << 16);
  return o;
}
__device__ __forceinline__ uint4 pack8f(const float* s) {
  uint4 o;
  o.x = (unsigned)f2bf(s[0]) | ((unsigned)f2bf(s[1]) << 16);
  o.y = (unsigned)f2bf(s[2]) | ((unsigned)f2bf(s[3]) << 16);
  o.z = (unsigned)f2bf(s[4]) | ((unsigned)f2bf(s[5]) << 16);
  o.w = (unsigned)f2bf(s[6]) | ((unsigned)f2bf(s[7]) << 16);
  return o;
}

// ---------------- group-norm statistics ----------------
// grid: 1024 blocks = 2 tensors * 16 batches * 32 groups; 256 threads
__global__ void gn_stats_kernel(const float* __restrict__ x,
                                const float* __restrict__ y,
                                float* __restrict__ stats) {
  int idx = blockIdx.x;
  int tensor = idx >> 9;
  int b = (idx >> 5) & 15;
  int g = idx & 31;
  const float* src = (tensor ? y : x) + ((size_t)b * CCH + (size_t)g * 8) * NPIX;
  const float4* vp = (const float4*)src;               // 8 ch * 4096 = 32768 floats = 8192 float4
  float s = 0.f, q = 0.f;
  for (int i = threadIdx.x; i < 8192; i += 256) {
    float4 v = vp[i];
    s += v.x + v.y + v.z + v.w;
    q += v.x * v.x + v.y * v.y + v.z * v.z + v.w * v.w;
  }
  __shared__ float rs[256], rq[256];
  rs[threadIdx.x] = s; rq[threadIdx.x] = q;
  __syncthreads();
  for (int st = 128; st > 0; st >>= 1) {
    if (threadIdx.x < st) { rs[threadIdx.x] += rs[threadIdx.x + st]; rq[threadIdx.x] += rq[threadIdx.x + st]; }
    __syncthreads();
  }
  if (threadIdx.x == 0) {
    float mean = rs[0] * (1.0f / 32768.0f);
    float var  = rq[0] * (1.0f / 32768.0f) - mean * mean;
    float rstd = rsqrtf(var + EPSV);
    stats[(size_t)idx * 2]     = mean;
    stats[(size_t)idx * 2 + 1] = rstd;
  }
}

// ---------------- normalize + transpose to [b][n][c] bf16 ----------------
// grid: (n tiles 64, c tiles 4, tensor*16+b = 32); 256 threads; 64x64 tile via LDS
__global__ void norm_t_kernel(const float* __restrict__ x,
                              const float* __restrict__ y,
                              const float* __restrict__ gn_w,
                              const float* __restrict__ gn_b,
                              const float* __restrict__ stats,
                              unsigned short* __restrict__ xt,
                              unsigned short* __restrict__ yt) {
  int tensor = blockIdx.z >> 4;
  int b      = blockIdx.z & 15;
  const float* src       = (tensor ? y : x) + (size_t)b * CCH * NPIX;
  unsigned short* dst    = (tensor ? yt : xt) + (size_t)b * NPIX * CCH;
  int c0 = blockIdx.y * 64, n0 = blockIdx.x * 64;
  __shared__ unsigned short lt[64][72];

  int t  = threadIdx.x;
  int cl = t >> 2;              // 0..63
  int nc = (t & 3) * 16;        // 0,16,32,48
  int c  = c0 + cl;
  int g  = c >> 3;
  int sidx = (tensor * 512 + b * 32 + g) * 2;
  float mean = stats[sidx], rstd = stats[sidx + 1];
  float sc = rstd * gn_w[c];
  float sh = gn_b[c] - mean * sc;
  const float4* rp = (const float4*)(src + (size_t)c * NPIX + n0 + nc);
#pragma unroll
  for (int j = 0; j < 4; ++j) {
    float4 v = rp[j];
    lt[cl][nc + j * 4 + 0] = f2bf(v.x * sc + sh);
    lt[cl][nc + j * 4 + 1] = f2bf(v.y * sc + sh);
    lt[cl][nc + j * 4 + 2] = f2bf(v.z * sc + sh);
    lt[cl][nc + j * 4 + 3] = f2bf(v.w * sc + sh);
  }
  __syncthreads();
  int nl = t >> 2;              // output row (n)
  int cc = (t & 3) * 16;        // output col chunk (c)
  uint4 p0, p1;
  p0.x = (unsigned)lt[cc + 0][nl]  | ((unsigned)lt[cc + 1][nl]  << 16);
  p0.y = (unsigned)lt[cc + 2][nl]  | ((unsigned)lt[cc + 3][nl]  << 16);
  p0.z = (unsigned)lt[cc + 4][nl]  | ((unsigned)lt[cc + 5][nl]  << 16);
  p0.w = (unsigned)lt[cc + 6][nl]  | ((unsigned)lt[cc + 7][nl]  << 16);
  p1.x = (unsigned)lt[cc + 8][nl]  | ((unsigned)lt[cc + 9][nl]  << 16);
  p1.y = (unsigned)lt[cc + 10][nl] | ((unsigned)lt[cc + 11][nl] << 16);
  p1.z = (unsigned)lt[cc + 12][nl] | ((unsigned)lt[cc + 13][nl] << 16);
  p1.w = (unsigned)lt[cc + 14][nl] | ((unsigned)lt[cc + 15][nl] << 16);
  unsigned short* drow = dst + (size_t)(n0 + nl) * CCH + c0 + cc;
  ((uint4*)drow)[0] = p0;
  ((uint4*)drow)[1] = p1;
}

// ---------------- f32 -> bf16 weight conversion ----------------
__global__ void cvt_bf16_kernel(const float* __restrict__ src,
                                unsigned short* __restrict__ dst, int n) {
  int i = blockIdx.x * 256 + threadIdx.x;
  if (i < n) dst[i] = f2bf(src[i]);
}

// ---------------- WMMA GEMM: D[o,n] = W[o,:] . actT[n,:] ----------------
// block: 128 threads = 4 waves; block tile 64(M) x 64(N); K = 256
// EPI 0: qkv pass1 (o<256 -> q1t transposed bf16; o>=256 -> k1 normal bf16)
// EPI 1: v2 normal bf16
// EPI 2: proj: f32 normal, fused  out = x + val + bias
template <int EPI>
__global__ void gemm_kernel(const unsigned short* __restrict__ W,      // [O][256] bf16
                            const unsigned short* __restrict__ actT,   // [B][NPIX][256] bf16
                            unsigned short* __restrict__ outTq,        // q1t [B][NPIX][256]
                            unsigned short* __restrict__ outNb,        // k1 or v2: [B][256][NPIX]
                            float* __restrict__ outF,                  // d_out
                            const float* __restrict__ xadd,
                            const float* __restrict__ bias) {
  const int K = CCH;
  __shared__ float tile[64][65];
  int lane = threadIdx.x & 31;
  int w    = threadIdx.x >> 5;
  int m15  = lane & 15;
  int h8   = (lane >> 4) * 8;
  int n0 = blockIdx.x * 64;
  int o0 = blockIdx.y * 64;
  int b  = blockIdx.z;
  const unsigned short* act   = actT + (size_t)b * NPIX * K;
  const unsigned short* wbase = W + (size_t)(o0 + w * 16 + m15) * K;

  v8f acc[4] = {};
  for (int kb = 0; kb < K; kb += 32) {
    Frag a;
    a.u[0] = *(const uint4*)(wbase + kb + h8);
    a.u[1] = *(const uint4*)(wbase + kb + 16 + h8);
#pragma unroll
    for (int nt = 0; nt < 4; ++nt) {
      Frag bfr;
      const unsigned short* arow = act + (size_t)(n0 + nt * 16 + m15) * K + kb + (h8 << 1);
      bfr.u[0] = *(const uint4*)(arow - h8);
      bfr.u[1] = *(const uint4*)(arow + 16 - h8);
      acc[nt] = __builtin_amdgcn_wmma_f32_16x16x32_bf16(
          false, a.v, false, bfr.v, (short)0, acc[nt], false, false);
    }
  }

  bool transposed = (EPI == 0) && (o0 < 256);
  if (transposed) {
    // D layout: lane holds column n, 8 consecutive M rows -> contiguous bf16 store
    int ob = o0 + w * 16 + (lane >> 4) * 8;
#pragma unroll
    for (int nt = 0; nt < 4; ++nt) {
      int n = n0 + nt * 16 + m15;
      *(uint4*)(outTq + ((size_t)b * NPIX + n) * CCH + ob) = pack8(acc[nt]);
    }
  } else {
    int rbase = w * 16 + (lane >> 4) * 8;
#pragma unroll
    for (int nt = 0; nt < 4; ++nt)
#pragma unroll
      for (int r = 0; r < 8; ++r)
        tile[rbase + r][nt * 16 + m15] = acc[nt][r];
    __syncthreads();
    int row = threadIdx.x >> 1;         // 0..63
    int cb  = (threadIdx.x & 1) * 32;   // 0 / 32
    if (EPI == 2) {
      int ch = o0 + row;
      size_t idx = ((size_t)b * CCH + ch) * NPIX + n0 + cb;
      float bv = bias[ch];
#pragma unroll
      for (int j = 0; j < 8; ++j) {
        float4 xv = ((const float4*)(xadd + idx))[j];
        float4 o;
        o.x = xv.x + tile[row][cb + j * 4 + 0] + bv;
        o.y = xv.y + tile[row][cb + j * 4 + 1] + bv;
        o.z = xv.z + tile[row][cb + j * 4 + 2] + bv;
        o.w = xv.w + tile[row][cb + j * 4 + 3] + bv;
        ((float4*)(outF + idx))[j] = o;
      }
    } else {
      int orow = (EPI == 0) ? (o0 - 256 + row) : (o0 + row);
      unsigned short* dr = outNb + ((size_t)b * CCH + orow) * NPIX + n0 + cb;
      uint4 q0, q1;
      q0.x = (unsigned)f2bf(tile[row][cb + 0])  | ((unsigned)f2bf(tile[row][cb + 1])  << 16);
      q0.y = (unsigned)f2bf(tile[row][cb + 2])  | ((unsigned)f2bf(tile[row][cb + 3])  << 16);
      q0.z = (unsigned)f2bf(tile[row][cb + 4])  | ((unsigned)f2bf(tile[row][cb + 5])  << 16);
      q0.w = (unsigned)f2bf(tile[row][cb + 6])  | ((unsigned)f2bf(tile[row][cb + 7])  << 16);
      q1.x = (unsigned)f2bf(tile[row][cb + 8])  | ((unsigned)f2bf(tile[row][cb + 9])  << 16);
      q1.y = (unsigned)f2bf(tile[row][cb + 10]) | ((unsigned)f2bf(tile[row][cb + 11]) << 16);
      q1.z = (unsigned)f2bf(tile[row][cb + 12]) | ((unsigned)f2bf(tile[row][cb + 13]) << 16);
      q1.w = (unsigned)f2bf(tile[row][cb + 14]) | ((unsigned)f2bf(tile[row][cb + 15]) << 16);
      ((uint4*)dr)[0] = q0;
      ((uint4*)dr)[1] = q1;
      uint4 r0, r1;
      r0.x = (unsigned)f2bf(tile[row][cb + 16]) | ((unsigned)f2bf(tile[row][cb + 17]) << 16);
      r0.y = (unsigned)f2bf(tile[row][cb + 18]) | ((unsigned)f2bf(tile[row][cb + 19]) << 16);
      r0.z = (unsigned)f2bf(tile[row][cb + 20]) | ((unsigned)f2bf(tile[row][cb + 21]) << 16);
      r0.w = (unsigned)f2bf(tile[row][cb + 22]) | ((unsigned)f2bf(tile[row][cb + 23]) << 16);
      r1.x = (unsigned)f2bf(tile[row][cb + 24]) | ((unsigned)f2bf(tile[row][cb + 25]) << 16);
      r1.y = (unsigned)f2bf(tile[row][cb + 26]) | ((unsigned)f2bf(tile[row][cb + 27]) << 16);
      r1.z = (unsigned)f2bf(tile[row][cb + 28]) | ((unsigned)f2bf(tile[row][cb + 29]) << 16);
      r1.w = (unsigned)f2bf(tile[row][cb + 30]) | ((unsigned)f2bf(tile[row][cb + 31]) << 16);
      ((uint4*)(dr + 16))[0] = r0;
      ((uint4*)(dr + 16))[1] = r1;
    }
  }
}

// ---------------- row softmax (in place, bf16, rows of 4096) ----------------
__global__ void softmax_kernel(unsigned short* __restrict__ k1) {
  int t = threadIdx.x;
  unsigned short* rowp = k1 + (size_t)blockIdx.x * NPIX;
  uint4 r0 = ((const uint4*)rowp)[t];
  uint4 r1 = ((const uint4*)rowp)[t + 256];
  float v[16];
  v[0]  = bf2f(r0.x & 0xffff); v[1]  = bf2f(r0.x >> 16);
  v[2]  = bf2f(r0.y & 0xffff); v[3]  = bf2f(r0.y >> 16);
  v[4]  = bf2f(r0.z & 0xffff); v[5]  = bf2f(r0.z >> 16);
  v[6]  = bf2f(r0.w & 0xffff); v[7]  = bf2f(r0.w >> 16);
  v[8]  = bf2f(r1.x & 0xffff); v[9]  = bf2f(r1.x >> 16);
  v[10] = bf2f(r1.y & 0xffff); v[11] = bf2f(r1.y >> 16);
  v[12] = bf2f(r1.z & 0xffff); v[13] = bf2f(r1.z >> 16);
  v[14] = bf2f(r1.w & 0xffff); v[15] = bf2f(r1.w >> 16);

  __shared__ float red[256];
  float m = -3.4e38f;
#pragma unroll
  for (int i = 0; i < 16; ++i) m = fmaxf(m, v[i]);
  red[t] = m; __syncthreads();
  for (int s = 128; s > 0; s >>= 1) {
    if (t < s) red[t] = fmaxf(red[t], red[t + s]);
    __syncthreads();
  }
  m = red[0]; __syncthreads();
  float sum = 0.f;
#pragma unroll
  for (int i = 0; i < 16; ++i) { v[i] = __expf(v[i] - m); sum += v[i]; }
  red[t] = sum; __syncthreads();
  for (int s = 128; s > 0; s >>= 1) {
    if (t < s) red[t] += red[t + s];
    __syncthreads();
  }
  float inv = 1.0f / red[0];
#pragma unroll
  for (int i = 0; i < 16; ++i) v[i] *= inv;
  uint4 o0, o1;
  o0.x = (unsigned)f2bf(v[0])  | ((unsigned)f2bf(v[1])  << 16);
  o0.y = (unsigned)f2bf(v[2])  | ((unsigned)f2bf(v[3])  << 16);
  o0.z = (unsigned)f2bf(v[4])  | ((unsigned)f2bf(v[5])  << 16);
  o0.w = (unsigned)f2bf(v[6])  | ((unsigned)f2bf(v[7])  << 16);
  o1.x = (unsigned)f2bf(v[8])  | ((unsigned)f2bf(v[9])  << 16);
  o1.y = (unsigned)f2bf(v[10]) | ((unsigned)f2bf(v[11]) << 16);
  o1.z = (unsigned)f2bf(v[12]) | ((unsigned)f2bf(v[13]) << 16);
  o1.w = (unsigned)f2bf(v[14]) | ((unsigned)f2bf(v[15]) << 16);
  ((uint4*)rowp)[t]       = o0;
  ((uint4*)rowp)[t + 256] = o1;
}

// ---------------- attention: attT[e,d] = sum_n k1s[d,n]*v2[e,n] ----------------
// grid: 128 blocks (b*h); 256 threads = 8 waves, K split across waves, LDS reduce
__global__ void att_kernel(const unsigned short* __restrict__ k1s,
                           const unsigned short* __restrict__ v2,
                           unsigned short* __restrict__ attT) {
  int bh = blockIdx.x;
  int b = bh >> 3, h = bh & 7;
  int lane = threadIdx.x & 31;
  int w    = threadIdx.x >> 5;
  int m15 = lane & 15;
  int h8  = (lane >> 4) * 8;
  const unsigned short* kb = k1s + ((size_t)b * CCH + h * DH) * NPIX;
  const unsigned short* vb = v2  + ((size_t)b * CCH + h * DH) * NPIX;

  v8f acc[2][2] = {};
  int k0 = w * 512;
  for (int kk = k0; kk < k0 + 512; kk += 32) {
    Frag a[2], bfr[2];
#pragma unroll
    for (int dt = 0; dt < 2; ++dt) {
      const unsigned short* p = kb + (size_t)(dt * 16 + m15) * NPIX + kk;
      a[dt].u[0] = *(const uint4*)(p + h8);
      a[dt].u[1] = *(const uint4*)(p + 16 + h8);
    }
#pragma unroll
    for (int et = 0; et < 2; ++et) {
      const unsigned short* p = vb + (size_t)(et * 16 + m15) * NPIX + kk + (h8 << 1);
      bfr[et].u[0] = *(const uint4*)(p - h8);
      bfr[et].u[1] = *(const uint4*)(p + 16 - h8);
    }
#pragma unroll
    for (int dt = 0; dt < 2; ++dt)
#pragma unroll
      for (int et = 0; et < 2; ++et)
        acc[dt][et] = __builtin_amdgcn_wmma_f32_16x16x32_bf16(
            false, a[dt].v, false, bfr[et].v, (short)0, acc[dt][et], false, false);
  }

  __shared__ float red[8][4][32][8];   // wave, tile, lane, r
#pragma unroll
  for (int dt = 0; dt < 2; ++dt)
#pragma unroll
    for (int et = 0; et < 2; ++et)
#pragma unroll
      for (int r = 0; r < 8; ++r)
        red[w][dt * 2 + et][lane][r] = acc[dt][et][r];
  __syncthreads();
  if (w < 4) {
    int dt = w >> 1, et = w & 1;
    float s[8];
#pragma unroll
    for (int r = 0; r < 8; ++r) s[r] = 0.f;
    for (int ww = 0; ww < 8; ++ww)
#pragma unroll
      for (int r = 0; r < 8; ++r) s[r] += red[ww][w][lane][r];
    // D layout: lane column e, 8 consecutive d rows -> contiguous attT[e][d] store
    int e  = et * 16 + m15;
    int d0 = dt * 16 + (lane >> 4) * 8;
    *(uint4*)(attT + ((size_t)bh * 32 + e) * 32 + d0) = pack8f(s);
  }
}

// ---------------- out: outT[n][h*32+e] = sum_d attT[e,d]*q1t[n][h*32+d] ----------------
// grid: (32, 128); 256 threads = 8 waves; wave handles one 16-wide n tile (K=32 -> 1 WMMA/tile)
__global__ void outg_kernel(const unsigned short* __restrict__ attT,
                            const unsigned short* __restrict__ q1t,
                            unsigned short* __restrict__ outT) {
  int bh = blockIdx.y;
  int b = bh >> 3, h = bh & 7;
  int lane = threadIdx.x & 31;
  int w    = threadIdx.x >> 5;
  int m15 = lane & 15;
  int h8  = (lane >> 4) * 8;
  int n0 = (blockIdx.x * 8 + w) * 16;
  const unsigned short* attb = attT + (size_t)bh * 32 * 32;

  Frag a[2];
#pragma unroll
  for (int et = 0; et < 2; ++et) {
    const unsigned short* p = attb + (size_t)(et * 16 + m15) * 32;
    a[et].u[0] = *(const uint4*)(p + h8);
    a[et].u[1] = *(const uint4*)(p + 16 + h8);
  }
  Frag bfr;
  int n = n0 + m15;
  const unsigned short* qp = q1t + ((size_t)b * NPIX + n) * CCH + h * DH + (h8 << 1);
  bfr.u[0] = *(const uint4*)(qp - h8);
  bfr.u[1] = *(const uint4*)(qp + 16 - h8);

#pragma unroll
  for (int et = 0; et < 2; ++et) {
    v8f acc = {};
    acc = __builtin_amdgcn_wmma_f32_16x16x32_bf16(
        false, a[et].v, false, bfr.v, (short)0, acc, false, false);
    int c = h * DH + et * 16 + (lane >> 4) * 8;
    *(uint4*)(outT + ((size_t)b * NPIX + n) * CCH + c) = pack8(acc);
  }
}

// ---------------- host launcher ----------------
extern "C" void kernel_launch(void* const* d_in, const int* in_sizes, int n_in,
                              void* d_out, int out_size, void* d_ws, size_t ws_size,
                              hipStream_t stream) {
  (void)in_sizes; (void)n_in; (void)out_size; (void)ws_size;
  const float* x      = (const float*)d_in[0];
  const float* y      = (const float*)d_in[1];
  const float* gn_w   = (const float*)d_in[2];
  const float* gn_b   = (const float*)d_in[3];
  const float* qkv1_w = (const float*)d_in[4];
  const float* qkv2_w = (const float*)d_in[5];
  const float* proj_w = (const float*)d_in[6];
  const float* proj_b = (const float*)d_in[7];
  float* out = (float*)d_out;

  char* ws = (char*)d_ws;
  const size_t SZ_ACT = (size_t)BATCH * NPIX * CCH * sizeof(unsigned short); // 32 MB
  size_t o_stats = 0;                                  // 2*16*32*2 f32 = 8 KB
  size_t o_wqk   = 8192;                               // 512*256 bf16
  size_t o_wv    = o_wqk + (size_t)512 * 256 * 2;      // 256*256 bf16
  size_t o_wp    = o_wv  + (size_t)256 * 256 * 2;      // 256*256 bf16
  size_t o_xt    = o_wp  + (size_t)256 * 256 * 2;      // [B][N][C] bf16
  size_t o_yt    = o_xt  + SZ_ACT;
  size_t o_q1t   = o_yt  + SZ_ACT;
  size_t o_k1    = o_q1t + SZ_ACT;
  size_t o_v2    = o_k1  + SZ_ACT;
  size_t o_att   = o_v2  + SZ_ACT;                     // 128*32*32 bf16 = 256 KB
  size_t o_outT  = o_xt;                               // alias: xt dead after qkv pass1

  float*          stats = (float*)(ws + o_stats);
  unsigned short* wqk   = (unsigned short*)(ws + o_wqk);
  unsigned short* wv    = (unsigned short*)(ws + o_wv);
  unsigned short* wp    = (unsigned short*)(ws + o_wp);
  unsigned short* xt    = (unsigned short*)(ws + o_xt);
  unsigned short* yt    = (unsigned short*)(ws + o_yt);
  unsigned short* q1t   = (unsigned short*)(ws + o_q1t);
  unsigned short* k1    = (unsigned short*)(ws + o_k1);
  unsigned short* v2    = (unsigned short*)(ws + o_v2);
  unsigned short* attT  = (unsigned short*)(ws + o_att);
  unsigned short* outT  = (unsigned short*)(ws + o_outT);

  // 1. group-norm stats for x and y
  gn_stats_kernel<<<1024, 256, 0, stream>>>(x, y, stats);
  // 2. normalize + transpose -> bf16 [b][n][c]
  norm_t_kernel<<<dim3(64, 4, 32), 256, 0, stream>>>(x, y, gn_w, gn_b, stats, xt, yt);
  // 3. weight conversion (only the rows actually used)
  cvt_bf16_kernel<<<512, 256, 0, stream>>>(qkv1_w, wqk, 512 * 256);             // q,k rows
  cvt_bf16_kernel<<<256, 256, 0, stream>>>(qkv2_w + 512 * 256, wv, 256 * 256);  // v rows
  cvt_bf16_kernel<<<256, 256, 0, stream>>>(proj_w, wp, 256 * 256);
  // 4. qkv pass1: q -> q1t (transposed bf16), k -> k1 (normal bf16)
  gemm_kernel<0><<<dim3(64, 8, BATCH), 128, 0, stream>>>(wqk, xt, q1t, k1, nullptr, nullptr, nullptr);
  // 5. qkv pass2: v2 (normal bf16)
  gemm_kernel<1><<<dim3(64, 4, BATCH), 128, 0, stream>>>(wv, yt, nullptr, v2, nullptr, nullptr, nullptr);
  // 6. softmax over n on k1 rows (in place)
  softmax_kernel<<<BATCH * CCH, 256, 0, stream>>>(k1);
  // 7. att = k1s . v2^T  (K = 4096 split across 8 waves, LDS-reduced), stored transposed
  att_kernel<<<BATCH * NH, 256, 0, stream>>>(k1, v2, attT);
  // 8. out = att^T . q1  -> outT [b][n][c] bf16
  outg_kernel<<<dim3(32, BATCH * NH), 256, 0, stream>>>(attT, q1t, outT);
  // 9. proj + residual: d_out = x + W_proj . out + b
  gemm_kernel<2><<<dim3(64, 4, BATCH), 128, 0, stream>>>(wp, outT, nullptr, nullptr, out, x, proj_b);
}
